// Convolution_vect_10866267259002
// MI455X (gfx1250) — matrix operand order repaired and checked by reading, not compiled, not run on hardware
//
#include <hip/hip_runtime.h>

// ---- CDNA5 (gfx1250) wave32 WMMA types ----
typedef __attribute__((ext_vector_type(16))) __bf16 v16bf;
typedef __attribute__((ext_vector_type(8)))  __bf16 v8bf;
typedef __attribute__((ext_vector_type(8)))  float  v8f;

#define NUM_ATOMS  512
#define NUM_FEATS  96
#define NUM_IRREPS 48
#define NK3   1270
#define NK3P  1280          // padded to multiple of 32 for bf16 WMMA K-steps
#define NK3F  1248          // end of the fully-in-bounds K-steps = K-tail start (39 * 32)
#define NK2   24
#define NK1   3
#define XYI   (48*48*48)    // 110592
#define XY    (48*48)       // 2304

// ---------------- prep kernels (tiny, atom-independent weighting) ----------------

__global__ void prep_w3t_kernel(const float* __restrict__ w3, __bf16* __restrict__ w3t) {
    int idx = blockIdx.x * blockDim.x + threadIdx.x;       // over 96*1280
    if (idx >= NUM_FEATS * NK3P) return;
    int c = idx / NK3P, k = idx % NK3P;
    float v = (k < NK3) ? w3[(size_t)k * NUM_FEATS + c] : 0.0f;
    w3t[idx] = (__bf16)v;
}

__global__ void prep_u2w_kernel(const float* __restrict__ U2, const float* __restrict__ w2,
                                float* __restrict__ U2W) {
    int idx = blockIdx.x * blockDim.x + threadIdx.x;       // over 96*2304
    if (idx >= NUM_FEATS * XY) return;
    int c = idx / XY, xy = idx % XY;
    float s = 0.0f;
#pragma unroll
    for (int k = 0; k < NK2; ++k)
        s += U2[(size_t)xy * NK2 + k] * w2[k * NUM_FEATS + c];
    U2W[idx] = s;
}

__global__ void prep_u1w_kernel(const float* __restrict__ U1, const float* __restrict__ w1,
                                float* __restrict__ U1W) {
    int idx = blockIdx.x * blockDim.x + threadIdx.x;       // over 96*48
    if (idx >= NUM_FEATS * 48) return;
    int c = idx / 48, x = idx % 48;
    float s = 0.0f;
#pragma unroll
    for (int k = 0; k < NK1; ++k)
        s += U1[x * NK1 + k] * w1[k * NUM_FEATS + c];
    U1W[idx] = s;
}

// ---------------- phase 1: U3w[c][xyi] = sum_k U3[xyi,k] * w3[k,c]  (bf16 WMMA) ----------------
// Tiles: M = c (16), N = xyi (16), K = 1280 in steps of 32.  A = w3^T (bf16, padded),
// B = U3 rows converted f32->bf16 in flight.  562 MB of U3 streamed once -> memory bound.
// Main loop is fully unguarded; the K-tail (k=1248..1269) is a single branchless
// epilogue using clamped addresses + select (no EXEC-mask ladders in the hot loop).

__device__ __forceinline__ v16bf load_a_frag(const __bf16* __restrict__ arow, int kk) {
    // A fragment: element j -> K=kk+16*half+j
    return *(const v16bf*)(arow + kk);
}

__global__ __launch_bounds__(256)
void u3w_gemm_kernel(const float* __restrict__ U3, const __bf16* __restrict__ w3t,
                     __bf16* __restrict__ U3W) {
    const int tid  = threadIdx.x;
    const int wave = tid >> 5;
    const int lane = tid & 31;
    const int half = lane >> 4;
    const int l16  = lane & 15;
    const int m0 = blockIdx.x * 128 + wave * 16;   // xyi tile (8 waves/block)
    const int c0 = blockIdx.y * 16;                // channel tile

    const float*  brow = U3  + (size_t)(m0 + l16) * NK3;           // B: col n = xyi
    const __bf16* arow = w3t + (size_t)(c0 + l16) * NK3P + 16*half; // A: row m = c

    v8f acc = {0.f,0.f,0.f,0.f,0.f,0.f,0.f,0.f};

    // ---- 39 fully in-bounds K-steps ----
    for (int kk = 0; kk < NK3F; kk += 32) {
        v16bf a = load_a_frag(arow, kk);
        // B fragment: element j -> K = kk + 16*half + j   (all in bounds)
        const float* bp = brow + kk + 16*half;
        v16bf b;
#pragma unroll
        for (int j = 0; j < 16; ++j) b[j] = (__bf16)bp[j];
        acc = __builtin_amdgcn_wmma_f32_16x16x32_bf16(false, a, false, b,
                                                      (short)0, acc, false, false);
    }

    // ---- branchless K-tail: kk = 1248, valid k < 1270 ----
    {
        v16bf a = load_a_frag(arow, NK3F);         // w3t is zero-padded to 1280
        const int kb = NK3F + 16*half;
        v16bf b;
#pragma unroll
        for (int j = 0; j < 16; ++j) {
            const int idx = kb + j;
            const int cl  = idx < NK3 ? idx : (NK3 - 1);   // clamp: load always legal
            float v = brow[cl];
            b[j] = (idx < NK3) ? (__bf16)v : (__bf16)0.0f; // v_cndmask, no branches
        }
        acc = __builtin_amdgcn_wmma_f32_16x16x32_bf16(false, a, false, b,
                                                      (short)0, acc, false, false);
    }

    // D: row M = c0 + r + 8*half, col N = m0 + l16 ; store bf16 coalesced along xyi
#pragma unroll
    for (int r = 0; r < 8; ++r) {
        const int cc = c0 + r + 8*half;
        U3W[(size_t)cc * XYI + m0 + l16] = (__bf16)acc[r];
    }
}

// ---------------- phase 2: fused per-atom contraction ----------------
// Block = (c, 16-atom tile), 4 waves x 12 x-values each.
// P[y,b] = sum_i U3w_c[x,y,i]*nf[b,i] via 2x WMMA (K=48 padded to 64),
// then out2 += (P + U2w)*nf[b,y]  (VALU on D-fragment), half-wave shfl reduce,
// final += (out2 + U1w)*nf[b,x], LDS ds_add_f32 across waves.

__global__ __launch_bounds__(128)
void contract_kernel(const float* __restrict__ nf, const __bf16* __restrict__ U3W,
                     const float* __restrict__ U2W, const float* __restrict__ U1W,
                     float* __restrict__ out) {
    __shared__ __bf16 nfb[16][64];    // bf16 node feats, i padded 48->64 with zeros
    __shared__ float  nff[16][48];    // f32 copy for the cheap VALU steps
    __shared__ float  u2w[XY];
    __shared__ float  u1w[48];
    __shared__ float  accum[16];

    const int c   = blockIdx.y;
    const int b0  = blockIdx.x * 16;
    const int tid = threadIdx.x;

    for (int idx = tid; idx < 16 * 64; idx += 128) {
        const int bl = idx >> 6, i = idx & 63;
        float v = 0.0f;
        if (i < 48) {
            v = nf[((size_t)(b0 + bl) * NUM_FEATS + c) * NUM_IRREPS + i];
            nff[bl][i] = v;
        }
        nfb[bl][i] = (__bf16)v;
    }
    for (int idx = tid; idx < XY; idx += 128) u2w[idx] = U2W[(size_t)c * XY + idx];
    if (tid < 48) u1w[tid] = U1W[c * 48 + tid];
    if (tid < 16) accum[tid] = 0.0f;
    __syncthreads();

    const int wave = tid >> 5, lane = tid & 31, half = lane >> 4, l16 = lane & 15;
    const __bf16* A = U3W + (size_t)c * XYI;   // [x][y][i] bf16, L2-resident (21 MB total)
    float facc = 0.0f;

    for (int x = wave * 12; x < wave * 12 + 12; ++x) {
        float o2 = 0.0f;                        // partial of out2[b,x] over this lane's y's
        for (int yt = 0; yt < 3; ++yt) {
            const int y0 = yt * 16;
            const __bf16* ar = A + (size_t)(x * 48 + y0 + l16) * 48;  // A row = y
            v8bf a1lo = *(const v8bf*)(ar + 8*half);          // i = 8*half .. +8
            v8bf a1hi = *(const v8bf*)(ar + 16 + 8*half);     // i = 16+8*half .. +8
            v8bf a2lo = *(const v8bf*)(ar + 32 + 8*half);     // i = 32+8*half .. +8
            v16bf a1, a2;
#pragma unroll
            for (int j = 0; j < 8; ++j) {
                a1[j] = a1lo[j];  a1[j+8] = a1hi[j];
                a2[j] = a2lo[j];  a2[j+8] = (__bf16)0.0f;     // i >= 48 pad
            }
            v8bf b1lo = *(const v8bf*)(&nfb[l16][8 * half]);
            v8bf b1hi = *(const v8bf*)(&nfb[l16][16 + 8 * half]);
            v8bf b2lo = *(const v8bf*)(&nfb[l16][32 + 8 * half]);
            v8bf b2hi = *(const v8bf*)(&nfb[l16][48 + 8 * half]);  // zeros beyond i=48
            v16bf b1, b2;
#pragma unroll
            for (int j = 0; j < 8; ++j) {
                b1[j] = b1lo[j];  b1[j+8] = b1hi[j];
                b2[j] = b2lo[j];  b2[j+8] = b2hi[j];
            }
            v8f p = {0.f,0.f,0.f,0.f,0.f,0.f,0.f,0.f};
            p = __builtin_amdgcn_wmma_f32_16x16x32_bf16(false, a1, false, b1,
                                                        (short)0, p, false, false);
            p = __builtin_amdgcn_wmma_f32_16x16x32_bf16(false, a2, false, b2,
                                                        (short)0, p, false, false);
            // D: row y = y0 + r + 8*half, col b = l16
#pragma unroll
            for (int r = 0; r < 8; ++r) {
                const int y = y0 + r + 8*half;
                o2 += (p[r] + u2w[x * 48 + y]) * nff[l16][y];
            }
        }
        o2 += __shfl_xor(o2, 16, 32);           // sum the two half-wave y-partitions
        if (half == 0)
            facc += (o2 + u1w[x]) * nff[l16][x];
    }
    if (half == 0) atomicAdd(&accum[l16], facc);  // ds_add_f32 across 4 waves
    __syncthreads();
    if (tid < 16) out[(size_t)(b0 + tid) * NUM_FEATS + c] = accum[tid];
}

// ---------------- launch ----------------

extern "C" void kernel_launch(void* const* d_in, const int* in_sizes, int n_in,
                              void* d_out, int out_size, void* d_ws, size_t ws_size,
                              hipStream_t stream) {
    (void)in_sizes; (void)n_in; (void)out_size; (void)ws_size;
    const float* nf = (const float*)d_in[0];
    const float* w3 = (const float*)d_in[1];
    const float* w2 = (const float*)d_in[2];
    const float* w1 = (const float*)d_in[3];
    const float* U3 = (const float*)d_in[4];
    const float* U2 = (const float*)d_in[5];
    const float* U1 = (const float*)d_in[6];
    float* out = (float*)d_out;

    char* ws = (char*)d_ws;
    __bf16* w3t = (__bf16*)ws;                                   // 96*1280*2  = 245760 B
    float*  U2W = (float*)(ws + 245760);                         // 96*2304*4  = 884736 B
    float*  U1W = (float*)(ws + 245760 + 884736);                // 96*48*4    = 18432 B
    __bf16* U3W = (__bf16*)(ws + 245760 + 884736 + 18432);       // 96*110592*2 = 21233664 B

    prep_w3t_kernel<<<(NUM_FEATS * NK3P + 255) / 256, 256, 0, stream>>>(w3, w3t);
    prep_u2w_kernel<<<(NUM_FEATS * XY + 255) / 256, 256, 0, stream>>>(U2, w2, U2W);
    prep_u1w_kernel<<<(NUM_FEATS * 48 + 255) / 256, 256, 0, stream>>>(U1, w1, U1W);

    u3w_gemm_kernel<<<dim3(XYI / 128, NUM_FEATS / 16), 256, 0, stream>>>(U3, w3t, U3W);
    contract_kernel<<<dim3(NUM_ATOMS / 16, NUM_FEATS), 128, 0, stream>>>(nf, U3W, U2W, U1W, out);
}
